// MessageModule_72524817760780
// MI455X (gfx1250) — matrix operand order repaired and checked by reading, not compiled, not run on hardware
//
#include <hip/hip_runtime.h>
#include <hip/hip_bf16.h>

// ---------------------------------------------------------------------------
// MI455X (gfx1250) implementation of MessageModule.
// Matmuls run on v_wmma_f32_16x16x32_bf16 (bf16 operands, f32 accumulate):
// the 16x16x32 shape gives 8x the FLOPs/instr of f32 16x16x4 WMMA, which is
// what moves this ~130 GFLOP / 1.5 GB workload from compute-bound to the
// 23.3 TB/s HBM roofline. Elementwise math stays in f32.
// ---------------------------------------------------------------------------

typedef __attribute__((ext_vector_type(16))) __bf16 bf16x16;
typedef __attribute__((ext_vector_type(8)))  float  v8f;
typedef __bf16 bf16_t;

#define PI_OVER_CUTOFF 0.314159265358979f
#define NEG_SLOPE 0.01f

// ---------------- WMMA fragment helpers ------------------------------------
// A fragment: 16x32 bf16 from a row-major f32 LDS tile (row stride ld).
// Layout (ISA 7.12.2, 16-bit A 16x32): lanes 0-15 hold K=k0..k0+7 and
// k0+16..k0+23; lanes 16-31 hold K=k0+8..15 and k0+24..31 (M = lane&15).
__device__ __forceinline__ bf16x16 loadA(const float* T, int ld, int k0) {
  const int lane = threadIdx.x & 31;
  const float* r = T + (lane & 15) * ld + k0 + ((lane >> 4) << 3);
  float4 f0 = *(const float4*)(r);
  float4 f1 = *(const float4*)(r + 4);
  float4 f2 = *(const float4*)(r + 16);
  float4 f3 = *(const float4*)(r + 20);
  bf16x16 a;
  a[0]=(bf16_t)f0.x; a[1]=(bf16_t)f0.y; a[2]=(bf16_t)f0.z; a[3]=(bf16_t)f0.w;
  a[4]=(bf16_t)f1.x; a[5]=(bf16_t)f1.y; a[6]=(bf16_t)f1.z; a[7]=(bf16_t)f1.w;
  a[8]=(bf16_t)f2.x; a[9]=(bf16_t)f2.y; a[10]=(bf16_t)f2.z; a[11]=(bf16_t)f2.w;
  a[12]=(bf16_t)f3.x; a[13]=(bf16_t)f3.y; a[14]=(bf16_t)f3.z; a[15]=(bf16_t)f3.w;
  return a;
}

// B fragment: pre-swizzled by the pack kernel; one 32B contiguous load/lane.
__device__ __forceinline__ bf16x16 loadB(const bf16_t* W, int frag) {
  return *(const bf16x16*)(W + (frag << 9) + ((threadIdx.x & 31) << 4));
}

// D scatter: 16x16 f32 C/D layout — lane holds col (lane&15), rows
// (lane<16 ? 0..7 : 8..15). MODE 2: C = C * D (fused gating multiply).
template<int MODE>
__device__ __forceinline__ void storeD(float* T, int ld, int n0, v8f d) {
  const int lane = threadIdx.x & 31;
  float* p = T + ((lane >> 4) << 3) * ld + n0 + (lane & 15);
#pragma unroll
  for (int j = 0; j < 8; ++j) {
    if constexpr (MODE == 2) p[j * ld] *= d[j];
    else                     p[j * ld]  = d[j];
  }
}

// C[16][N] (LDS) (=|*=) A[16][K] (LDS) * W^T (+bias). MODE: 0 store,
// 1 bias+store, 2 bias then multiply-into-C.
template<int K, int N, int MODE>
__device__ __forceinline__ void mm(const float* A, int ldA, float* C, int ldC,
                                   const bf16_t* W, const float* bias) {
  constexpr int nKb = K / 32;
#pragma unroll
  for (int nb = 0; nb < N / 16; ++nb) {
    v8f acc = {};
#pragma unroll
    for (int kb = 0; kb < nKb; ++kb) {
      bf16x16 a = loadA(A, ldA, kb * 32);
      bf16x16 b = loadB(W, nb * nKb + kb);
      acc = __builtin_amdgcn_wmma_f32_16x16x32_bf16(
          false, a, false, b, (short)0, acc, false, false);
    }
    if constexpr (MODE >= 1) {
      float bb = bias[nb * 16 + (threadIdx.x & 15)];
#pragma unroll
      for (int j = 0; j < 8; ++j) acc[j] += bb;
    }
    storeD<(MODE == 2) ? 2 : 0>(C, ldC, nb * 16, acc);
  }
}

// ---------------- weight packing --------------------------------------------
struct PackJob { const float* src; int dstOff; int N; int K; };
struct PackArgs { PackJob j[17]; };

__global__ void pack_weights(PackArgs a, bf16_t* dst) {
  PackJob p = a.j[blockIdx.x];
  const int total = p.N * p.K;
  const int nKb = p.K >> 5;
  for (int t = threadIdx.x; t < total; t += blockDim.x) {
    int frag = t >> 9, within = t & 511;
    int lane = within >> 4, jj = within & 15;
    int nb = frag / nKb, kb = frag - nb * nKb;
    int row = nb * 16 + (lane & 15);
    int col = kb * 32 + ((lane & 16) ? 8 : 0) + (jj & 7) + ((jj & 8) ? 16 : 0);
    dst[p.dstOff + t] = (bf16_t)p.src[row * p.K + col];
  }
}

// ---------------- node precompute -------------------------------------------
struct NodeArgs {
  const float *nsca, *nvec;
  const bf16_t *wv1, *wv2, *ws, *wg;
  const float *bg;
  float *ns_ws, *nv_ws;
  int N;
};

__global__ __launch_bounds__(32) void node_kernel(NodeArgs A) {
  __shared__ __align__(16) float sm[8192];
  float* V  = sm;          // [3][16][32] input vec; reused for out_v
  float* VI = sm + 1536;   // [3][16][32] hidden vec
  float* S  = sm + 3072;   // [16][160] concat(norm, sca)
  float* OS = sm + 5632;   // [16][128] out_s
  float* G  = sm + 7680;   // [16][32] gate
  const int lane = threadIdx.x;
  const int n0 = blockIdx.x * 16;
  if (n0 >= A.N) return;

  for (int t = lane; t < 1536; t += 32) {          // vec -> coord-major LDS
    int m = t / 96, r = t - m * 96, i = r / 3, c = r - i * 3;
    int nm = n0 + m; if (nm >= A.N) nm = A.N - 1;
    V[c * 512 + m * 32 + i] = A.nvec[(size_t)nm * 96 + r];
  }
  for (int t = lane; t < 2048; t += 32) {          // sca -> S[:,32:160]
    int m = t >> 7, k = t & 127;
    int nm = n0 + m; if (nm >= A.N) nm = A.N - 1;
    S[m * 160 + 32 + k] = A.nsca[(size_t)nm * 128 + k];
  }
  for (int c = 0; c < 3; ++c)                      // vi = v @ Wv1^T
    mm<32, 32, 0>(V + c * 512, 32, VI + c * 512, 32, A.wv1, nullptr);
  for (int t = lane; t < 512; t += 32) {           // norm -> S[:,0:32]
    int m = t >> 5, h = t & 31;
    float a = VI[m * 32 + h], b = VI[512 + m * 32 + h], c2 = VI[1024 + m * 32 + h];
    S[m * 160 + h] = sqrtf(a * a + b * b + c2 * c2);
  }
  mm<160, 128, 0>(S, 160, OS, 128, A.ws, nullptr); // out_s
  mm<128, 32, 1>(OS, 128, G, 32, A.wg, A.bg);      // gate pre-act
  for (int t = lane; t < 512; t += 32) G[t] = 1.f / (1.f + __expf(-G[t]));
  for (int c = 0; c < 3; ++c)                      // out_v = vi @ Wv2^T
    mm<32, 32, 0>(VI + c * 512, 32, V + c * 512, 32, A.wv2, nullptr);
  for (int t = lane; t < 512; t += 32) {           // gate out_v
    float g = G[t]; V[t] *= g; V[512 + t] *= g; V[1024 + t] *= g;
  }
  for (int t = lane; t < 2048; t += 32) {          // write ns_
    int m = t >> 7, k = t & 127; int nm = n0 + m;
    if (nm < A.N) A.ns_ws[(size_t)nm * 128 + k] = OS[m * 128 + k];
  }
  for (int t = lane; t < 1536; t += 32) {          // write nv_ as [n][c*32+i]
    int m = t / 96, r = t - m * 96, c = r >> 5, i = r & 31; int nm = n0 + m;
    if (nm < A.N) A.nv_ws[(size_t)nm * 96 + r] = V[c * 512 + m * 32 + i];
  }
}

// ---------------- edge megakernel -------------------------------------------
struct EdgeArgs {
  const float *esca, *evec; const int* eidx; const float* dist;
  const float *ns_ws, *nv_ws;
  const bf16_t *wv1, *wv2, *ws, *wg, *wd;
  const bf16_t *wsca, *we2n, *wn2e, *wevn;
  const bf16_t *owv1, *owv2, *ows, *owg;
  const float *ebg, *scab, *e2nb, *n2eb, *evnb, *obg;
  float *out_s, *out_v;
  int E;
};

__global__ __launch_bounds__(32) void edge_kernel(EdgeArgs A) {
  __shared__ __align__(16) float sm[14864];
  float* V  = sm;           // [3][16][32] edge vec; later y_v
  float* VI = sm + 1536;    // [3][16][32] hidden; later vi2
  float* S  = sm + 3072;    // [16][96] then [16][160] concat buffers
  float* ES = sm + 5632;    // [16][64] edge out_s / es_
  float* G  = sm + 6656;    // [16][32] gates
  float* EV = sm + 7168;    // [3][16][32] ev_; later out_v
  float* DD = sm + 8704;    // [3][16][32] VN d; later ev_lin
  float* NS = sm + 10240;   // [16][128] ns_g -> y_s -> final out_s
  float* NV = sm + 12288;   // [16][96] nv_g (row-major, [m][c*32+o])
  float* T1 = sm + 13824;   // [16][32]
  float* T2 = sm + 14336;   // [16][32]
  float* CC = sm + 14848;   // [16] cutoff
  const int lane = threadIdx.x;
  const int e0 = blockIdx.x * 16;
  if (e0 >= A.E) return;

  // ---- stage inputs
  for (int t = lane; t < 1536; t += 32) {          // edge vec -> coord-major
    int m = t / 96, r = t - m * 96, i = r / 3, c = r - i * 3;
    int em = e0 + m; if (em >= A.E) em = A.E - 1;
    V[c * 512 + m * 32 + i] = A.evec[(size_t)em * 96 + r];
  }
  for (int t = lane; t < 1024; t += 32) {          // edge sca -> S[:,32:96]
    int m = t >> 6, k = t & 63;
    int em = e0 + m; if (em >= A.E) em = A.E - 1;
    S[m * 96 + 32 + k] = A.esca[(size_t)em * 64 + k];
  }
  if (lane < 16) {                                 // annealing cutoff
    int em = e0 + lane; if (em >= A.E) em = A.E - 1;
    float d = A.dist[em];
    float c = 0.5f * (__cosf(d * PI_OVER_CUTOFF) + 1.f);
    CC[lane] = (d <= 10.f && d >= 0.f) ? c : 0.f;
  }
  for (int r = 0; r < 16; ++r) {                   // gather node features (L2)
    int em = e0 + r; if (em >= A.E) em = A.E - 1;
    int idx = A.eidx[em];
    ((float4*)(NS + r * 128))[lane] =
        ((const float4*)(A.ns_ws + (size_t)idx * 128))[lane];
    if (lane < 24)
      ((float4*)(NV + r * 96))[lane] =
          ((const float4*)(A.nv_ws + (size_t)idx * 96))[lane];
  }

  // ---- edge GVLinear
  for (int c = 0; c < 3; ++c)
    mm<32, 32, 0>(V + c * 512, 32, VI + c * 512, 32, A.wv1, nullptr);
  for (int t = lane; t < 512; t += 32) {           // norm -> S[:,0:32]
    int m = t >> 5, h = t & 31;
    float a = VI[m * 32 + h], b = VI[512 + m * 32 + h], c2 = VI[1024 + m * 32 + h];
    S[m * 96 + h] = sqrtf(a * a + b * b + c2 * c2);
  }
  mm<96, 64, 0>(S, 96, ES, 64, A.ws, nullptr);     // es_raw
  mm<64, 32, 1>(ES, 64, G, 32, A.wg, A.ebg);       // gate
  for (int t = lane; t < 512; t += 32) G[t] = 1.f / (1.f + __expf(-G[t]));
  for (int c = 0; c < 3; ++c)
    mm<32, 32, 0>(VI + c * 512, 32, EV + c * 512, 32, A.wv2, nullptr);
  for (int t = lane; t < 512; t += 32) {           // gate ev_
    float g = G[t]; EV[t] *= g; EV[512 + t] *= g; EV[1024 + t] *= g;
  }

  // ---- VN-LeakyReLU on ev_
  for (int c = 0; c < 3; ++c)
    mm<32, 32, 0>(EV + c * 512, 32, DD + c * 512, 32, A.wd, nullptr);
  for (int t = lane; t < 512; t += 32) {
    float x0 = EV[t], x1 = EV[512 + t], x2 = EV[1024 + t];
    float d0 = DD[t], d1 = DD[512 + t], d2 = DD[1024 + t];
    float dot = x0 * d0 + x1 * d1 + x2 * d2;
    if (dot < 0.f) {
      float s = (1.f - NEG_SLOPE) * dot / (d0 * d0 + d1 * d1 + d2 * d2 + 1e-6f);
      EV[t] = x0 - s * d0; EV[512 + t] = x1 - s * d1; EV[1024 + t] = x2 - s * d2;
    } // dot>=0: result == x (ns*x + (1-ns)*x)
  }
  for (int t = lane; t < 1024; t += 32) {          // leaky on es_
    float x = ES[t]; ES[t] = x >= 0.f ? x : NEG_SLOPE * x;
  }

  // ---- combine
  mm<128, 32, 1>(NS, 128, T2, 32, A.wn2e, A.n2eb); // ns_g @ n2e (before NS reuse)
  mm<64, 32, 1>(ES, 64, T1, 32, A.we2n, A.e2nb);   // es_ @ e2n
  mm<64, 128, 2>(ES, 64, NS, 128, A.wsca, A.scab); // y_s = ns_g * (es@W + b)
  for (int c = 0; c < 3; ++c)                      // ev_lin (into DD)
    mm<32, 32, 1>(EV + c * 512, 32, DD + c * 512, 32, A.wevn, A.evnb);
  for (int t = lane; t < 512; t += 32) {           // y_v (into V)
    int m = t >> 5, o = t & 31;
    float a = T1[t], b = T2[t];
    V[t]        = a * NV[m * 96 + o]      + b * DD[t];
    V[512 + t]  = a * NV[m * 96 + 32 + o] + b * DD[512 + t];
    V[1024 + t] = a * NV[m * 96 + 64 + o] + b * DD[1024 + t];
  }

  // ---- output GVLinear
  for (int c = 0; c < 3; ++c)
    mm<32, 32, 0>(V + c * 512, 32, VI + c * 512, 32, A.owv1, nullptr);
  for (int t = lane; t < 512; t += 32) {           // norm2 -> S[:,0:32] (ld 160)
    int m = t >> 5, h = t & 31;
    float a = VI[m * 32 + h], b = VI[512 + m * 32 + h], c2 = VI[1024 + m * 32 + h];
    S[m * 160 + h] = sqrtf(a * a + b * b + c2 * c2);
  }
  for (int t = lane; t < 2048; t += 32) {          // y_s -> S[:,32:160]
    int m = t >> 7, k = t & 127;
    S[m * 160 + 32 + k] = NS[m * 128 + k];
  }
  mm<160, 128, 0>(S, 160, NS, 128, A.ows, nullptr); // out_s (NS reused)
  mm<128, 32, 1>(NS, 128, G, 32, A.owg, A.obg);     // gate2
  for (int t = lane; t < 512; t += 32) G[t] = 1.f / (1.f + __expf(-G[t]));
  for (int c = 0; c < 3; ++c)                       // out_v (EV reused)
    mm<32, 32, 0>(VI + c * 512, 32, EV + c * 512, 32, A.owv2, nullptr);
  for (int t = lane; t < 512; t += 32) {            // gate2 * C
    int m = t >> 5; float f = G[t] * CC[m];
    EV[t] *= f; EV[512 + t] *= f; EV[1024 + t] *= f;
  }

  // ---- writeout (coalesced)
  for (int t = lane; t < 2048; t += 32) {
    int m = t >> 7, k = t & 127; int em = e0 + m;
    if (em < A.E) A.out_s[(size_t)em * 128 + k] = NS[m * 128 + k] * CC[m];
  }
  for (int t = lane; t < 1536; t += 32) {
    int m = t / 96, r = t - m * 96, i = r / 3, c = r - i * 3; int em = e0 + m;
    if (em < A.E) A.out_v[(size_t)em * 96 + r] = EV[c * 512 + m * 32 + i];
  }
}

// ---------------- host launch ------------------------------------------------
// packed-weight offsets (bf16 elements)
#define O_nWv1 0
#define O_nWv2 1024
#define O_nWs  2048
#define O_nWg  22528
#define O_eWv1 26624
#define O_eWv2 27648
#define O_eWs  28672
#define O_eWg  34816
#define O_eWd  36864
#define O_sca  37888
#define O_e2n  46080
#define O_n2e  48128
#define O_evn  52224
#define O_oWv1 53248
#define O_oWv2 54272
#define O_oWs  55296
#define O_oWg  75776

extern "C" void kernel_launch(void* const* d_in, const int* in_sizes, int n_in,
                              void* d_out, int out_size, void* d_ws, size_t ws_size,
                              hipStream_t stream) {
  const int N = in_sizes[0] / 128;
  const int E = in_sizes[4];

  float* ns_ws = (float*)d_ws;                        // [N][128]
  float* nv_ws = ns_ws + (size_t)N * 128;             // [N][96] coord-major rows
  bf16_t* wpk  = (bf16_t*)(nv_ws + (size_t)N * 96);   // packed bf16 weights

  PackArgs pa;
  auto J = [&](int s, int din, int off, int n, int k) {
    pa.j[s] = PackJob{(const float*)d_in[din], off, n, k};
  };
  J(0, 6,  O_nWv1, 32, 32);  J(1, 7,  O_nWv2, 32, 32);
  J(2, 8,  O_nWs, 128, 160); J(3, 9,  O_nWg, 32, 128);
  J(4, 11, O_eWv1, 32, 32);  J(5, 12, O_eWv2, 32, 32);
  J(6, 13, O_eWs, 64, 96);   J(7, 14, O_eWg, 32, 64);
  J(8, 16, O_eWd, 32, 32);
  J(9, 17, O_sca, 128, 64);  J(10, 19, O_e2n, 32, 64);
  J(11, 21, O_n2e, 32, 128); J(12, 23, O_evn, 32, 32);
  J(13, 25, O_oWv1, 32, 32); J(14, 26, O_oWv2, 32, 32);
  J(15, 27, O_oWs, 128, 160);J(16, 28, O_oWg, 32, 128);
  pack_weights<<<17, 256, 0, stream>>>(pa, wpk);

  NodeArgs na;
  na.nsca = (const float*)d_in[0]; na.nvec = (const float*)d_in[1];
  na.wv1 = wpk + O_nWv1; na.wv2 = wpk + O_nWv2;
  na.ws = wpk + O_nWs;   na.wg = wpk + O_nWg;
  na.bg = (const float*)d_in[10];
  na.ns_ws = ns_ws; na.nv_ws = nv_ws; na.N = N;
  node_kernel<<<(N + 15) / 16, 32, 0, stream>>>(na);

  EdgeArgs ea;
  ea.esca = (const float*)d_in[2]; ea.evec = (const float*)d_in[3];
  ea.eidx = (const int*)d_in[4];   ea.dist = (const float*)d_in[5];
  ea.ns_ws = ns_ws; ea.nv_ws = nv_ws;
  ea.wv1 = wpk + O_eWv1; ea.wv2 = wpk + O_eWv2; ea.ws = wpk + O_eWs;
  ea.wg = wpk + O_eWg;   ea.wd = wpk + O_eWd;
  ea.wsca = wpk + O_sca; ea.we2n = wpk + O_e2n;
  ea.wn2e = wpk + O_n2e; ea.wevn = wpk + O_evn;
  ea.owv1 = wpk + O_oWv1; ea.owv2 = wpk + O_oWv2;
  ea.ows = wpk + O_oWs;   ea.owg = wpk + O_oWg;
  ea.ebg = (const float*)d_in[15];
  ea.scab = (const float*)d_in[18]; ea.e2nb = (const float*)d_in[20];
  ea.n2eb = (const float*)d_in[22]; ea.evnb = (const float*)d_in[24];
  ea.obg = (const float*)d_in[29];
  ea.out_s = (float*)d_out;
  ea.out_v = (float*)d_out + (size_t)E * 128;
  ea.E = E;
  edge_kernel<<<(E + 15) / 16, 32, 0, stream>>>(ea);
}
